// SPNet_17411797418341
// MI455X (gfx1250) — matrix operand order repaired and checked by reading, not compiled, not run on hardware
//
#include <hip/hip_runtime.h>

// ---------------------------------------------------------------------------
// SPNet forward on MI455X (gfx1250, wave32).
// GEMMs: bf16 WMMA (v_wmma_f32_16x16x32_bf16), fp32 accumulate.
// Sparse: wave-per-edge gathers + global_atomic_add_f32 scatters.
// ---------------------------------------------------------------------------

typedef __bf16 bf16;
typedef __attribute__((ext_vector_type(16))) __bf16 v16bf;
typedef __attribute__((ext_vector_type(8)))  float  v8f;

#define THREADS 256

__device__ __forceinline__ float lrelu02(float v) { return v > 0.f ? v : 0.2f * v; }

// float atomic max via int/uint ordering trick (works with 0xFF..FF init).
__device__ __forceinline__ void atomicMaxF(float* addr, float val) {
  if (val >= 0.f) atomicMax((int*)addr, __float_as_int(val));
  else            atomicMin((unsigned int*)addr, __float_as_uint(val));
}

// ---------------- degree / norm ----------------
__global__ void k_degree(const int* __restrict__ dst, int* __restrict__ degi, int E) {
  int i = blockIdx.x * blockDim.x + threadIdx.x;
  if (i < E) atomicAdd(&degi[dst[i]], 1);
}

__global__ void k_dinv(const int* __restrict__ degi, float* __restrict__ dinv, int n) {
  int i = blockIdx.x * blockDim.x + threadIdx.x;
  if (i < n) dinv[i] = rsqrtf((float)(degi[i] + 1));  // +1 self loop
}

// ---------------- WMMA fragment packing ----------------
// B fragment order (32x16 bf16 tile): lane l -> col n=l%16, kblock=l/16;
// element j -> k = 32*kc + 16*(j/8) + 8*kblock + j%8.   W is [K,128] row-major.
__global__ void k_pack_b(const float* __restrict__ W, bf16* __restrict__ dst, int K) {
  int total = K * 128;
  for (int o = blockIdx.x * blockDim.x + threadIdx.x; o < total;
       o += gridDim.x * blockDim.x) {
    int j = o & 15;
    int t = o >> 4;
    int l = t & 31; t >>= 5;
    int nt = t & 7;
    int kc = t >> 3;
    int n = nt * 16 + (l & 15);
    int k = kc * 32 + ((j >> 3) << 4) + ((l >> 4) << 3) + (j & 7);
    dst[o] = (bf16)W[k * 128 + n];
  }
}

// A fragment order (16x32 bf16 tile, ISA 7.12.2 16-bit A table):
// lane l -> row m = rt*16 + l%16, half = l/16;
// element j: v=j/2, kb = v<4 ? 2v : 16+2(v-4); k = 32*kc + kb + 8*half + (j&1).
// A1 non-null => concatenated [A0 | A1] input of width 256.
__global__ void k_pack_a(const float* __restrict__ A0, const float* __restrict__ A1,
                         bf16* __restrict__ dst, int nRows, int K) {
  size_t total = (size_t)nRows * K;
  int kChunks = K >> 5;
  for (size_t o = (size_t)blockIdx.x * blockDim.x + threadIdx.x; o < total;
       o += (size_t)gridDim.x * blockDim.x) {
    int j = (int)(o & 15);
    size_t t = o >> 4;
    int l = (int)(t & 31); t >>= 5;
    int kc = (int)(t % kChunks);
    size_t rt = t / kChunks;
    int m = (int)rt * 16 + (l & 15);
    int half = l >> 4;
    int v = j >> 1;
    int kb = (v < 4) ? (2 * v) : (16 + 2 * (v - 4));
    int k = kc * 32 + kb + 8 * half + (j & 1);
    float val = (k < 128) ? A0[(size_t)m * 128 + k]
                          : A1[(size_t)m * 128 + (k - 128)];
    dst[o] = (bf16)val;
  }
}

// ---------------- WMMA GEMM: C[N,128] = act(A[N,K] @ W[K,128] + bias) ------
// 8 waves/block, each wave owns a 16-row tile across all 128 cols.
template <int KC>  // KC = K/32 (4 or 8), compile-time so frags stay in VGPRs
__global__ void k_wmma_gemm(const bf16* __restrict__ Afrag,
                            const bf16* __restrict__ Bfrag,
                            const float* __restrict__ bias,
                            float* __restrict__ C, int rowTiles, int act) {
  int wave = threadIdx.x >> 5;
  int lane = threadIdx.x & 31;
  int rt = blockIdx.x * 8 + wave;
  if (rt >= rowTiles) return;
  const v16bf* A16 = (const v16bf*)Afrag + (size_t)rt * KC * 32;
  const v16bf* B16 = (const v16bf*)Bfrag;
  v16bf a[KC];
#pragma unroll
  for (int kc = 0; kc < KC; ++kc) a[kc] = A16[kc * 32 + lane];
  int n0 = lane & 15;
  int m0 = rt * 16 + ((lane >> 4) << 3);  // D layout: lanes16-31 hold M=8..15
#pragma unroll
  for (int nt = 0; nt < 8; ++nt) {
    v8f acc = {};
#pragma unroll
    for (int kc = 0; kc < KC; ++kc) {
      v16bf b = B16[(kc * 8 + nt) * 32 + lane];
      acc = __builtin_amdgcn_wmma_f32_16x16x32_bf16(
          false, a[kc], false, b, (short)0, acc, false, false);
    }
    int n = nt * 16 + n0;
    float bv = bias ? bias[n] : 0.f;
#pragma unroll
    for (int r = 0; r < 8; ++r) {
      float vv = acc[r] + bv;
      if (act) vv = lrelu02(vv);
      C[(size_t)(m0 + r) * 128 + n] = vv;
    }
  }
}

// ---------------- GCN neighbor scatter (wave per edge) ----------------
__global__ void k_gcn_scatter(const int* __restrict__ src, const int* __restrict__ dst,
                              const float* __restrict__ dinv,
                              const float* __restrict__ xwo, const float* __restrict__ xwt,
                              float* __restrict__ aggo, float* __restrict__ aggt, int E) {
  int lane = threadIdx.x & 31;
  int wid = (blockIdx.x * blockDim.x + threadIdx.x) >> 5;
  int nw = (gridDim.x * blockDim.x) >> 5;
  for (int e = wid; e < E; e += nw) {
    int s = src[e], d = dst[e];
    float nm = dinv[s] * dinv[d];
    const float4* o4 = (const float4*)(xwo + (size_t)s * 128);
    const float4* t4 = (const float4*)(xwt + (size_t)s * 128);
    __builtin_prefetch(aggo + (size_t)d * 128, 1, 0);
    float4 vo = o4[lane], vt = t4[lane];
    float* po = aggo + (size_t)d * 128 + lane * 4;
    float* pt = aggt + (size_t)d * 128 + lane * 4;
    atomicAdd(po + 0, vo.x * nm); atomicAdd(po + 1, vo.y * nm);
    atomicAdd(po + 2, vo.z * nm); atomicAdd(po + 3, vo.w * nm);
    atomicAdd(pt + 0, vt.x * nm); atomicAdd(pt + 1, vt.y * nm);
    atomicAdd(pt + 2, vt.z * nm); atomicAdd(pt + 3, vt.w * nm);
  }
}

// r = relu(agg + self_loop_term + bias), done in place for both GCN branches.
__global__ void k_gcn_finalize(float* __restrict__ ro, float* __restrict__ rt,
                               const float* __restrict__ xwo, const float* __restrict__ xwt,
                               const float* __restrict__ dinv,
                               const float* __restrict__ bo, const float* __restrict__ bt,
                               size_t total) {
  for (size_t i = (size_t)blockIdx.x * blockDim.x + threadIdx.x; i < total;
       i += (size_t)gridDim.x * blockDim.x) {
    int col = (int)(i & 127);
    size_t row = i >> 7;
    float di = dinv[row];
    float sl = di * di;  // self-loop norm
    float vo = ro[i] + xwo[i] * sl + bo[col];
    float vt = rt[i] + xwt[i] * sl + bt[col];
    ro[i] = vo > 0.f ? vo : 0.f;
    rt[i] = vt > 0.f ? vt : 0.f;
  }
}

// ---------------- attention ----------------
// e[i] = dot(r_o[i], a_w[:128]) + dot(r_t[i], a_w[128:]) + a_b   (wave/node)
__global__ void k_attn_e(const float* __restrict__ ro, const float* __restrict__ rt,
                         const float* __restrict__ aw, const float* __restrict__ ab,
                         float* __restrict__ e, int n) {
  int wid = (blockIdx.x * blockDim.x + threadIdx.x) >> 5;
  int lane = threadIdx.x & 31;
  if (wid >= n) return;
  const float4* r4 = (const float4*)(ro + (size_t)wid * 128);
  const float4* t4 = (const float4*)(rt + (size_t)wid * 128);
  const float4* a4 = (const float4*)aw;
  float4 rv = r4[lane], tv = t4[lane];
  float4 a0 = a4[lane], a1 = a4[lane + 32];
  float s = rv.x * a0.x + rv.y * a0.y + rv.z * a0.z + rv.w * a0.w +
            tv.x * a1.x + tv.y * a1.y + tv.z * a1.z + tv.w * a1.w;
#pragma unroll
  for (int off = 16; off; off >>= 1) s += __shfl_xor(s, off, 32);
  if (lane == 0) e[wid] = s + ab[0];
}

__global__ void k_attn_logits(const int* __restrict__ src, const int* __restrict__ dst,
                              const float* __restrict__ e, float* __restrict__ lbuf,
                              float* __restrict__ m, int E) {
  int i = blockIdx.x * blockDim.x + threadIdx.x;
  if (i >= E) return;
  float l = e[src[i]] + e[dst[i]];
  l = l > 0.f ? l : 0.2f * l;
  lbuf[i] = l;
  atomicMaxF(m + dst[i], l);
}

__global__ void k_attn_wsum(const int* __restrict__ src, const int* __restrict__ dst,
                            const int* __restrict__ t, const float* __restrict__ m,
                            float* __restrict__ wbuf, float* __restrict__ denom, int E) {
  int i = blockIdx.x * blockDim.x + threadIdx.x;
  if (i >= E) return;
  int d = dst[i];
  float mm = m[d];
  if (!(mm >= -3.4e38f && mm <= 3.4e38f)) mm = 0.f;  // where(isfinite(m), m, 0)
  float w = expf(wbuf[i] - mm) * (t[src[i]] > 0 ? 1.f : 0.f);
  wbuf[i] = w;
  if (w != 0.f) atomicAdd(denom + d, w);
}

__global__ void k_attn_scatter(const int* __restrict__ src, const int* __restrict__ dst,
                               const float* __restrict__ rt, const float* __restrict__ wbuf,
                               const float* __restrict__ denom, float* __restrict__ h, int E) {
  int lane = threadIdx.x & 31;
  int wid = (blockIdx.x * blockDim.x + threadIdx.x) >> 5;
  int nw = (gridDim.x * blockDim.x) >> 5;
  for (int e = wid; e < E; e += nw) {
    int d = dst[e];
    float alpha = wbuf[e] / (denom[d] + 1e-16f);
    if (alpha != 0.f) {
      int s = src[e];
      const float4* r4 = (const float4*)(rt + (size_t)s * 128);
      float4 v = r4[lane];
      float* hp = h + (size_t)d * 128 + lane * 4;
      atomicAdd(hp + 0, v.x * alpha); atomicAdd(hp + 1, v.y * alpha);
      atomicAdd(hp + 2, v.z * alpha); atomicAdd(hp + 3, v.w * alpha);
    }
  }
}

// ---------------- head final dot: out[i] = X[i,:] . w + b  (wave/node) -----
__global__ void k_dot_head(const float* __restrict__ X, const float* __restrict__ w,
                           const float* __restrict__ b, float* __restrict__ out, int n) {
  int wid = (blockIdx.x * blockDim.x + threadIdx.x) >> 5;
  int lane = threadIdx.x & 31;
  if (wid >= n) return;
  const float4* x4 = (const float4*)(X + (size_t)wid * 128);
  const float4* w4 = (const float4*)w;
  float4 xv = x4[lane], wv = w4[lane];
  float s = xv.x * wv.x + xv.y * wv.y + xv.z * wv.z + xv.w * wv.w;
#pragma unroll
  for (int off = 16; off; off >>= 1) s += __shfl_xor(s, off, 32);
  if (lane == 0) out[wid] = s + b[0];
}

// pred_t = sigmoid(sd); pred = t>0 ? s1 : s0
__global__ void k_final(const int* __restrict__ t, const float* __restrict__ s1,
                        const float* __restrict__ s0, const float* __restrict__ sd,
                        float* __restrict__ out, int n) {
  int i = blockIdx.x * blockDim.x + threadIdx.x;
  if (i >= n) return;
  out[i] = 1.f / (1.f + expf(-sd[i]));
  out[n + i] = (t[i] > 0) ? s1[i] : s0[i];
}

// ---------------------------------------------------------------------------
extern "C" void kernel_launch(void* const* d_in, const int* in_sizes, int n_in,
                              void* d_out, int out_size, void* d_ws, size_t ws_size,
                              hipStream_t stream) {
  const int N = in_sizes[1];
  const int E = in_sizes[3] / 2;
  const int H = 128;

  // inputs (setup_inputs order)
  const float* x    = (const float*)d_in[0];
  const int*   t    = (const int*)d_in[1];
  const int*   ei   = (const int*)d_in[3];
  const int*   src  = ei;
  const int*   dst  = ei + E;
  const float* W_o  = (const float*)d_in[4];
  const float* b_o  = (const float*)d_in[5];
  const float* W_t  = (const float*)d_in[6];
  const float* b_t  = (const float*)d_in[7];
  const float* a_w  = (const float*)d_in[8];
  const float* a_b  = (const float*)d_in[9];
  const float* W_enc = (const float*)d_in[10];
  const float* b_enc = (const float*)d_in[11];
  const float* hW[3][2] = {{(const float*)d_in[12], (const float*)d_in[14]},
                           {(const float*)d_in[18], (const float*)d_in[20]},
                           {(const float*)d_in[24], (const float*)d_in[26]}};
  const float* hB[3][2] = {{(const float*)d_in[13], (const float*)d_in[15]},
                           {(const float*)d_in[19], (const float*)d_in[21]},
                           {(const float*)d_in[25], (const float*)d_in[27]}};
  const float* hW3[3] = {(const float*)d_in[16], (const float*)d_in[22], (const float*)d_in[28]};
  const float* hB3[3] = {(const float*)d_in[17], (const float*)d_in[23], (const float*)d_in[29]};

  float* out = (float*)d_out;
  float* z_enc = out + 2 * (size_t)N;  // [N,128] slice of output

  // ---- workspace layout (with reuse) ----
  char* W = (char*)d_ws;
  size_t off = 0;
  auto alloc = [&](size_t sz) { size_t o = off; off = (off + sz + 255) & ~(size_t)255; return o; };
  const size_t NH4 = (size_t)N * H * 4;  // fp32 [N,128]
  const size_t NH2 = (size_t)N * H * 2;  // bf16 [N,128]
  float* xw_o = (float*)(W + alloc(NH4));   // later reused: rt_bf (bf16, fits)
  float* xw_t = (float*)(W + alloc(NH4));   // later reused: cat_bf (N*256 bf16 == NH4)
  float* ro   = (float*)(W + alloc(NH4));   // agg_o -> r_o; later h1f32
  float* rt   = (float*)(W + alloc(NH4));   // agg_t -> r_t; later h2f32
  float* hbuf = (float*)(W + alloc(NH4));   // attention output h
  bf16*  xbf  = (bf16*)(W + alloc(NH2));    // x bf16 frags; later z_enc frags
  bf16*  h1bf = (bf16*)(W + alloc(NH2));    // head hidden frags
  int*   degi = (int*)(W + alloc((size_t)N * 4));
  float* dinv = (float*)(W + alloc((size_t)N * 4));
  float* evec = (float*)(W + alloc((size_t)N * 4));
  float* mvec = (float*)(W + alloc((size_t)N * 4));
  float* den  = (float*)(W + alloc((size_t)N * 4));
  float* s1   = (float*)(W + alloc((size_t)N * 4));
  float* s0   = (float*)(W + alloc((size_t)N * 4));
  float* sd   = (float*)(W + alloc((size_t)N * 4));
  float* lbuf = (float*)(W + alloc((size_t)E * 4));  // logits, then softmax w
  bf16* Bo    = (bf16*)(W + alloc(128 * 128 * 2));
  bf16* Bt    = (bf16*)(W + alloc(128 * 128 * 2));
  bf16* Benc  = (bf16*)(W + alloc(256 * 128 * 2));
  bf16* Bh[3][2];
  for (int p = 0; p < 3; ++p)
    for (int l = 0; l < 2; ++l) Bh[p][l] = (bf16*)(W + alloc(128 * 128 * 2));
  bf16* rt_bf  = (bf16*)xw_o;
  bf16* cat_bf = (bf16*)xw_t;
  bf16* z_bf   = xbf;
  float* h1f32 = ro;
  float* h2f32 = rt;

  const int rowTiles = (N + 15) / 16;           // N=100000 -> exact
  const int gemmBlocks = (rowTiles + 7) / 8;
  const int edgeWaveBlocks = (E + 7) / 8;       // wave per edge, 8 waves/block
  const int nodeWaveBlocks = (N + 7) / 8;
  size_t NHtot = (size_t)N * H;

  // ---- 0: init ----
  hipMemsetAsync(degi, 0, (size_t)N * 4, stream);
  hipMemsetAsync(ro, 0, NH4, stream);
  hipMemsetAsync(rt, 0, NH4, stream);
  hipMemsetAsync(hbuf, 0, NH4, stream);
  hipMemsetAsync(den, 0, (size_t)N * 4, stream);
  hipMemsetAsync(mvec, 0xFF, (size_t)N * 4, stream);  // acts as -inf for atomicMaxF

  // ---- 1: degree + norm ----
  k_degree<<<(E + THREADS - 1) / THREADS, THREADS, 0, stream>>>(dst, degi, E);
  k_dinv<<<(N + THREADS - 1) / THREADS, THREADS, 0, stream>>>(degi, dinv, N);

  // ---- 2: pack weights to bf16 WMMA B-fragments ----
  k_pack_b<<<64, THREADS, 0, stream>>>(W_o, Bo, 128);
  k_pack_b<<<64, THREADS, 0, stream>>>(W_t, Bt, 128);
  k_pack_b<<<128, THREADS, 0, stream>>>(W_enc, Benc, 256);
  for (int p = 0; p < 3; ++p)
    for (int l = 0; l < 2; ++l)
      k_pack_b<<<64, THREADS, 0, stream>>>(hW[p][l], Bh[p][l], 128);

  // ---- 3: xw_o / xw_t = x @ W (WMMA) ----
  k_pack_a<<<(int)((NHtot + THREADS - 1) / THREADS), THREADS, 0, stream>>>(x, nullptr, xbf, N, 128);
  k_wmma_gemm<4><<<gemmBlocks, THREADS, 0, stream>>>(xbf, Bo, nullptr, xw_o, rowTiles, 0);
  k_wmma_gemm<4><<<gemmBlocks, THREADS, 0, stream>>>(xbf, Bt, nullptr, xw_t, rowTiles, 0);

  // ---- 4: GCN scatter + finalize ----
  k_gcn_scatter<<<edgeWaveBlocks, THREADS, 0, stream>>>(src, dst, dinv, xw_o, xw_t, ro, rt, E);
  k_gcn_finalize<<<(int)((NHtot + THREADS - 1) / THREADS), THREADS, 0, stream>>>(
      ro, rt, xw_o, xw_t, dinv, b_o, b_t, NHtot);

  // r_t bf16 frags for discriminator head (reuses dead xw_o region)
  k_pack_a<<<(int)((NHtot + THREADS - 1) / THREADS), THREADS, 0, stream>>>(rt, nullptr, rt_bf, N, 128);

  // ---- 5: masked attention ----
  k_attn_e<<<nodeWaveBlocks, THREADS, 0, stream>>>(ro, rt, a_w, a_b, evec, N);
  k_attn_logits<<<(E + THREADS - 1) / THREADS, THREADS, 0, stream>>>(src, dst, evec, lbuf, mvec, E);
  k_attn_wsum<<<(E + THREADS - 1) / THREADS, THREADS, 0, stream>>>(src, dst, t, mvec, lbuf, den, E);
  k_attn_scatter<<<edgeWaveBlocks, THREADS, 0, stream>>>(src, dst, rt, lbuf, den, hbuf, E);

  // ---- 6: z_enc = [r_o, h] @ W_enc + b_enc (WMMA, K=256) ----
  k_pack_a<<<(int)((2 * NHtot + THREADS - 1) / THREADS), THREADS, 0, stream>>>(ro, hbuf, cat_bf, N, 256);
  k_wmma_gemm<8><<<gemmBlocks, THREADS, 0, stream>>>(cat_bf, Benc, b_enc, z_enc, rowTiles, 0);

  // ---- 7: heads (p1, p0 from z_enc; discriminator from r_t) ----
  k_pack_a<<<(int)((NHtot + THREADS - 1) / THREADS), THREADS, 0, stream>>>(z_enc, nullptr, z_bf, N, 128);
  float* hs[3] = {s1, s0, sd};
  for (int p = 0; p < 3; ++p) {
    const bf16* A0 = (p == 2) ? rt_bf : z_bf;
    k_wmma_gemm<4><<<gemmBlocks, THREADS, 0, stream>>>(A0, Bh[p][0], hB[p][0], h1f32, rowTiles, 1);
    k_pack_a<<<(int)((NHtot + THREADS - 1) / THREADS), THREADS, 0, stream>>>(h1f32, nullptr, h1bf, N, 128);
    k_wmma_gemm<4><<<gemmBlocks, THREADS, 0, stream>>>(h1bf, Bh[p][1], hB[p][1], h2f32, rowTiles, 1);
    k_dot_head<<<nodeWaveBlocks, THREADS, 0, stream>>>(h2f32, hW3[p], hB3[p], hs[p], N);
  }

  // ---- 8: outputs ----
  k_final<<<(N + THREADS - 1) / THREADS, THREADS, 0, stream>>>(t, s1, s0, sd, out, N);
}
